// DeepMoEReasoner_56186762166282
// MI455X (gfx1250) — matrix operand order repaired and checked by reading, not compiled, not run on hardware
//
#include <hip/hip_runtime.h>
#include <hip/hip_bf16.h>
#include <math.h>

typedef __attribute__((ext_vector_type(16))) _Float16 v16h;
typedef __attribute__((ext_vector_type(8)))  _Float16 v8h;
typedef __attribute__((ext_vector_type(8)))  float    v8f;

#define B_  4
#define S_  1024
#define D_  512
#define H_  8
#define DH_ 64
#define I_  2048
#define E_  8
#define L_  4
#define C_  4
#define D2_ 256
#define T_  (B_*S_)
#define D3_ (3*D_)
#define EPS_ 1e-5f

#if defined(__has_builtin)
#if __has_builtin(__builtin_amdgcn_global_load_async_to_lds_b128)
#define ASYNC_LDS 1
#endif
#endif

#ifdef ASYNC_LDS
// Builtin signature (from hipcc diagnostic): param0 is
// '__attribute__((__vector_size__(4 * sizeof(int)))) int __device__ *'
// i.e. an int4-vector pointer in the global address space; convention matches
// __builtin_amdgcn_global_load_lds: (global src, lds dst, imm offset, imm cpol).
typedef int v4i_ __attribute__((__vector_size__(4 * sizeof(int))));
typedef __attribute__((address_space(1))) v4i_ v4i_g;
typedef __attribute__((address_space(3))) v4i_ v4i_l;
__device__ __forceinline__ void async_cp16(const _Float16* g, _Float16* l) {
  __builtin_amdgcn_global_load_async_to_lds_b128((v4i_g*)g, (v4i_l*)l, 0, 0);
}
__device__ __forceinline__ void async_wait0() {
  asm volatile("s_wait_asynccnt 0x0" ::: "memory");
}
#endif

// ---------------- block reductions (wave32) ----------------
__device__ __forceinline__ float blk_sum(float v, float* sm) {
  __syncthreads();
  const int lane = threadIdx.x & 31, w = threadIdx.x >> 5;
  #pragma unroll
  for (int o = 16; o > 0; o >>= 1) v += __shfl_down(v, o, 32);
  if (lane == 0) sm[w] = v;
  __syncthreads();
  if (w == 0) {
    float t = (lane < ((int)blockDim.x >> 5)) ? sm[lane] : 0.f;
    #pragma unroll
    for (int o = 4; o > 0; o >>= 1) t += __shfl_down(t, o, 32);
    if (lane == 0) sm[0] = t;
  }
  __syncthreads();
  return sm[0];
}
__device__ __forceinline__ float blk_max(float v, float* sm) {
  __syncthreads();
  const int lane = threadIdx.x & 31, w = threadIdx.x >> 5;
  #pragma unroll
  for (int o = 16; o > 0; o >>= 1) v = fmaxf(v, __shfl_down(v, o, 32));
  if (lane == 0) sm[w] = v;
  __syncthreads();
  if (w == 0) {
    float t = (lane < ((int)blockDim.x >> 5)) ? sm[lane] : -3.4e38f;
    #pragma unroll
    for (int o = 4; o > 0; o >>= 1) t = fmaxf(t, __shfl_down(t, o, 32));
    if (lane == 0) sm[0] = t;
  }
  __syncthreads();
  return sm[0];
}

// ---------------- utility kernels ----------------
__global__ void k_fill(float* __restrict__ p, long n, float v) {
  long i = (long)blockIdx.x * blockDim.x + threadIdx.x;
  if (i < n) p[i] = v;
}
__global__ void k_copy(float* __restrict__ dst, const float* __restrict__ src, long n) {
  long i = (long)blockIdx.x * blockDim.x + threadIdx.x;
  if (i < n) dst[i] = src[i];
}
// f32 (N,K) -> f16 (N,K)
__global__ void k_cvt(const float* __restrict__ in, _Float16* __restrict__ out, long n) {
  long i = (long)blockIdx.x * blockDim.x + threadIdx.x;
  if (i < n) out[i] = (_Float16)in[i];
}
// f32 (K,N) row-major -> f16 (N,K) row-major (transpose + convert)
__global__ void k_cvt_t(const float* __restrict__ in, _Float16* __restrict__ out,
                        int K, int N) {
  long id = (long)blockIdx.x * blockDim.x + threadIdx.x;
  if (id >= (long)K * N) return;
  const int k = (int)(id % K);
  const int n = (int)(id / K);
  out[(long)n * K + k] = (_Float16)in[(long)k * N + n];
}
// V transpose: QKV16 (T,3D) -> VT16 [b][h][n(64)][k(1024)]
__global__ void k_vt(const _Float16* __restrict__ qkv, _Float16* __restrict__ vt) {
  long id = (long)blockIdx.x * blockDim.x + threadIdx.x; // B*H*64*1024
  const int k = (int)(id & 1023);
  long r = id >> 10;
  const int n = (int)(r & 63); r >>= 6;
  const int z = (int)(r & 7);
  const int b = (int)(r >> 3);
  vt[id] = qkv[((long)(b * S_ + k)) * D3_ + 2 * D_ + z * DH_ + n];
}

// ---------------- LayerNorm: y = LN(x (+ res*rowmask)) * w + b, D=512 ----------------
template<typename TIN, typename TOUT>
__global__ __launch_bounds__(256) void k_layernorm(
    const TIN* __restrict__ x, const float* __restrict__ res,
    const float* __restrict__ rowmask, const float* __restrict__ w,
    const float* __restrict__ bb, TOUT* __restrict__ y)
{
  __shared__ float sm[8];
  const long row = blockIdx.x;
  const int tid = threadIdx.x;
  const TIN* xr = x + row * D_;
  float e0 = (float)xr[tid], e1 = (float)xr[tid + 256];
  if (res) {
    const float rm = rowmask ? rowmask[row] : 1.f;
    const float* rr = res + row * D_;
    e0 += rr[tid] * rm;
    e1 += rr[tid + 256] * rm;
  }
  const float mean = blk_sum(e0 + e1, sm) * (1.f / D_);
  const float d0 = e0 - mean, d1 = e1 - mean;
  const float var = blk_sum(d0 * d0 + d1 * d1, sm) * (1.f / D_);
  const float rs = rsqrtf(var + EPS_);
  y[row * D_ + tid]       = (TOUT)(d0 * rs * w[tid]       + bb[tid]);
  y[row * D_ + tid + 256] = (TOUT)(d1 * rs * w[tid + 256] + bb[tid + 256]);
}

// ---------------- row softmax (cols=1024): out16 = softmax(in*scl) ----------------
__global__ __launch_bounds__(256) void k_softmax1024(const float* __restrict__ in,
    _Float16* __restrict__ out, float scl)
{
  __shared__ float sm[8];
  const float* p = in + (long)blockIdx.x * 1024;
  _Float16* q = out + (long)blockIdx.x * 1024;
  const int tid = threadIdx.x;
  float v[4];
  float mx = -3.4e38f;
  #pragma unroll
  for (int i = 0; i < 4; ++i) { v[i] = p[tid + i * 256] * scl; mx = fmaxf(mx, v[i]); }
  mx = blk_max(mx, sm);
  float s = 0.f;
  #pragma unroll
  for (int i = 0; i < 4; ++i) { v[i] = __expf(v[i] - mx); s += v[i]; }
  s = blk_sum(s, sm);
  const float inv = 1.f / s;
  #pragma unroll
  for (int i = 0; i < 4; ++i) q[tid + i * 256] = (_Float16)(v[i] * inv);
}

// ---------------- WMMA GEMM (all-f16 operands) ----------------
// C[M,N] = epilogue(A[M,K] @ B^T + bias), B stored (N,K) row-major, k-contiguous.
// NF: n-fragments per wave (4 -> BN=128, 2 -> BN=64). BM=128 always.
// ACT: 0 none, 1 exact GELU.
// CMODE: 0 f16 store; 1 f32 store + resid add; 2 f32 += rowscale*val; 3 f32 store.
template<int NF, int ACT, int CMODE>
__global__ __launch_bounds__(256) void k_gemm(
    const _Float16* __restrict__ A, int lda, long bsA,
    const _Float16* __restrict__ Bm, int ldb, long bsB,
    void* __restrict__ Cv, int ldc, long bsC,
    const float* __restrict__ bias,
    const float* __restrict__ resid,
    const float* __restrict__ scale, int scaleStride,
    int K)
{
  constexpr int BN = 32 * NF;
  __shared__ __align__(16) _Float16 As[128][40];
  __shared__ __align__(16) _Float16 Bs[BN][40];
  const int tid  = threadIdx.x;
  const int lane = tid & 31;
  const int wave = tid >> 5;
  const int wm = wave & 3, wn = wave >> 2;        // 4 x 2 waves; wave tile 32 x (16*NF)
  const int bm = blockIdx.y * 128;
  const int bn = blockIdx.x * BN;
  const int z  = blockIdx.z;
  const _Float16* Ab = A  + (long)z * bsA;
  const _Float16* Bb = Bm + (long)z * bsB;

  const v8f zero8 = {0.f,0.f,0.f,0.f,0.f,0.f,0.f,0.f};
  v8f acc[2][NF];
  #pragma unroll
  for (int tm = 0; tm < 2; ++tm)
    #pragma unroll
    for (int tn = 0; tn < NF; ++tn) acc[tm][tn] = zero8;

  // fragment addressing (CDNA5 wave32 WMMA layouts)
  const int lane16 = lane & 15, half = lane >> 4;
  const int arow = wm * 32 + lane16;
  const int ah   = half * 8;
  const int bh   = half * 8;

  for (int k0 = 0; k0 < K; k0 += 32) {
    __syncthreads();
    // stage A tile 128x32 halves: 2 x b128 per thread
    #pragma unroll
    for (int i = 0; i < 2; ++i) {
      const int c = tid + i * 256;            // 0..511
      const int r = c >> 2, seg = c & 3;
      const _Float16* gp = Ab + (long)(bm + r) * lda + (k0 + seg * 8);
      _Float16* lp = &As[r][seg * 8];
#ifdef ASYNC_LDS
      async_cp16(gp, lp);
#else
      *(v8h*)lp = *(const v8h*)gp;
#endif
    }
    // stage B tile BNx32 halves: NF/2 x b128 per thread
    #pragma unroll
    for (int i = 0; i < NF / 2; ++i) {
      const int c = tid + i * 256;
      const int r = c >> 2, seg = c & 3;
      const _Float16* gp = Bb + (long)(bn + r) * ldb + (k0 + seg * 8);
      _Float16* lp = &Bs[r][seg * 8];
#ifdef ASYNC_LDS
      async_cp16(gp, lp);
#else
      *(v8h*)lp = *(const v8h*)gp;
#endif
    }
    if (k0 + 32 < K) { // gfx1250 global_prefetch_b8 on next A tile
      __builtin_prefetch(Ab + (long)(bm + (tid >> 1)) * lda + (k0 + 32 + (tid & 1) * 16), 0, 1);
    }
#ifdef ASYNC_LDS
    async_wait0();
#endif
    __syncthreads();

    v16h af[2], bf[NF];
    #pragma unroll
    for (int tm = 0; tm < 2; ++tm) {
      const _Float16* p = &As[arow + tm * 16][0];
      const v8h lo = *(const v8h*)(p + ah);       // e0..7 : k = half*8 + e
      const v8h hi = *(const v8h*)(p + 16 + ah);  // e8..15: k = 16 + half*8 + e
      v16h a;
      #pragma unroll
      for (int q = 0; q < 8; ++q) { a[q] = lo[q]; a[8 + q] = hi[q]; }
      af[tm] = a;
    }
    #pragma unroll
    for (int tn = 0; tn < NF; ++tn) {
      const _Float16* p = &Bs[wn * (16 * NF) + tn * 16 + lane16][0];
      const v8h lo = *(const v8h*)(p + bh);       // e0..7 : k = half*8 + e
      const v8h hi = *(const v8h*)(p + 16 + bh);  // e8..15: k = 16 + half*8 + e
      v16h b;
      #pragma unroll
      for (int q = 0; q < 8; ++q) { b[q] = lo[q]; b[8 + q] = hi[q]; }
      bf[tn] = b;
    }
    #pragma unroll
    for (int tm = 0; tm < 2; ++tm)
      #pragma unroll
      for (int tn = 0; tn < NF; ++tn)
        acc[tm][tn] = __builtin_amdgcn_wmma_f32_16x16x32_f16(
            false, af[tm], false, bf[tn], (short)0, acc[tm][tn], false, false);
  }

  // epilogue: C fragment row = 8*half+v, col = lane16
  #pragma unroll
  for (int tm = 0; tm < 2; ++tm)
    #pragma unroll
    for (int tn = 0; tn < NF; ++tn) {
      const int col   = bn + wn * (16 * NF) + tn * 16 + lane16;
      const int rbase = bm + wm * 32 + tm * 16 + half * 8;
      const float bv = bias ? bias[col] : 0.f;
      #pragma unroll
      for (int vv = 0; vv < 8; ++vv) {
        const int row = rbase + vv;
        float val = acc[tm][tn][vv] + bv;
        if (ACT == 1) val = 0.5f * val * (1.f + erff(val * 0.70710678118f));
        const long off = (long)row * ldc + col;
        if (CMODE == 0) {
          ((_Float16*)Cv + (long)z * bsC)[off] = (_Float16)val;
        } else if (CMODE == 1) {
          ((float*)Cv)[off] = val + resid[off];
        } else if (CMODE == 2) {
          ((float*)Cv)[off] += scale[(long)row * scaleStride] * val;
        } else {
          ((float*)Cv + (long)z * bsC)[off] = val;
        }
      }
    }
}

// ---------------- MoE gating: softmax over E=8, top-2, normalized combine ----------------
__global__ __launch_bounds__(256) void k_gate(const _Float16* __restrict__ h,
    const float* __restrict__ gw, float* __restrict__ comb, float* __restrict__ usage)
{
  const int t = blockIdx.x * 256 + threadIdx.x;
  if (t >= T_) return;
  const _Float16* hr = h + (long)t * D_;
  float lg[E_];
  float mx = -3.4e38f;
  #pragma unroll
  for (int e = 0; e < E_; ++e) {
    const float* wr = gw + e * D_;
    float s = 0.f;
    for (int d = 0; d < D_; ++d) s += (float)hr[d] * wr[d];
    lg[e] = s; mx = fmaxf(mx, s);
  }
  float den = 0.f;
  #pragma unroll
  for (int e = 0; e < E_; ++e) { lg[e] = __expf(lg[e] - mx); den += lg[e]; }
  const float inv = 1.f / den;
  #pragma unroll
  for (int e = 0; e < E_; ++e) lg[e] *= inv;
  int i1 = 0;
  #pragma unroll
  for (int e = 1; e < E_; ++e) if (lg[e] > lg[i1]) i1 = e;
  int i2 = (i1 == 0) ? 1 : 0;
  #pragma unroll
  for (int e = 0; e < E_; ++e) if (e != i1 && lg[e] > lg[i2]) i2 = e;
  const float vn = 1.f / (lg[i1] + lg[i2]);
  #pragma unroll
  for (int e = 0; e < E_; ++e)
    comb[(long)t * E_ + e] = (e == i1) ? lg[i1] * vn : ((e == i2) ? lg[i2] * vn : 0.f);
  atomicAdd(&usage[i1], 1.f);
  atomicAdd(&usage[i2], 1.f);
}

__global__ void k_aux(const float* __restrict__ usage, float* __restrict__ aux)
{
  if (threadIdx.x == 0) {
    float s = 0.f;
    for (int e = 0; e < E_; ++e) {
      const float u = usage[e] * (1.f / T_) - (1.f / E_);
      s += u * u;
    }
    aux[0] += s * (1.f / E_);
  }
}
__global__ void k_write_aux(const float* __restrict__ aux, float* __restrict__ out)
{
  if (threadIdx.x == 0) out[0] = aux[0] * (1.f / L_);
}

// ---------------- masked mean pool over S ----------------
__global__ __launch_bounds__(256) void k_pool(const _Float16* __restrict__ x,
    const float* __restrict__ mask, float* __restrict__ pooled)
{
  const int d = blockIdx.x * 256 + threadIdx.x;
  const int b = blockIdx.y;
  float num = 0.f, den = 0.f;
  for (int s = 0; s < S_; ++s) {
    const float m = mask[b * S_ + s];
    num += (float)x[((long)(b * S_ + s)) * D_ + d] * m;
    den += m;
  }
  pooled[b * D_ + d] = num / fmaxf(den, 1e-9f);
}

// ---------------- small FC: y[b*N+j] = act(dot(x[b], W[j]) + bias[j]) ----------------
template<int ACT> // 0 none, 2 tanh, 3 relu, 4 sigmoid
__global__ void k_small_fc(const float* __restrict__ x, const float* __restrict__ W,
                           const float* __restrict__ bias, float* __restrict__ y,
                           int N, int Kd, int total)
{
  const int i = blockIdx.x * blockDim.x + threadIdx.x;
  if (i >= total) return;
  const int b = i / N, j = i % N;
  const float* xr = x + (long)b * Kd;
  const float* wr = W + (long)j * Kd;
  float s = bias ? bias[j] : 0.f;
  for (int k = 0; k < Kd; ++k) s += xr[k] * wr[k];
  if (ACT == 2)      s = tanhf(s);
  else if (ACT == 3) s = fmaxf(s, 0.f);
  else if (ACT == 4) s = 1.f / (1.f + __expf(-s));
  y[i] = s;
}

// ---------------- host ----------------
extern "C" void kernel_launch(void* const* d_in, const int* in_sizes, int n_in,
                              void* d_out, int out_size, void* d_ws, size_t ws_size,
                              hipStream_t stream) {
  (void)in_sizes; (void)n_in; (void)out_size; (void)ws_size;
  const float* emb    = (const float*)d_in[0];
  const float* amask  = (const float*)d_in[1];
  const float* in_w   = (const float*)d_in[2];
  const float* in_b   = (const float*)d_in[3];
  const float* out_w  = (const float*)d_in[4];
  const float* out_b  = (const float*)d_in[5];
  const float* ln1_w  = (const float*)d_in[6];
  const float* ln1_b  = (const float*)d_in[7];
  const float* ln2_w  = (const float*)d_in[8];
  const float* ln2_b  = (const float*)d_in[9];
  const float* gate_w = (const float*)d_in[10];
  const float* e_w1   = (const float*)d_in[11];
  const float* e_b1   = (const float*)d_in[12];
  const float* e_w2   = (const float*)d_in[13];
  const float* e_b2   = (const float*)d_in[14];
  const float* mln_w  = (const float*)d_in[15];
  const float* mln_b  = (const float*)d_in[16];
  const float* fln_w  = (const float*)d_in[17];
  const float* fln_b  = (const float*)d_in[18];
  const float* pool_w = (const float*)d_in[19];
  const float* pool_b = (const float*)d_in[20];
  const float* cls_w  = (const float*)d_in[21];
  const float* cls_b  = (const float*)d_in[22];
  const float* cf_w1  = (const float*)d_in[23];
  const float* cf_b1  = (const float*)d_in[24];
  const float* cf_w2  = (const float*)d_in[25];
  const float* cf_b2  = (const float*)d_in[26];
  float* out = (float*)d_out;

  // ---- workspace layout: f32 region, then f16 region (all chunks multiple of 8 elems) ----
  float* X   = (float*)d_ws;               // (T,D) residual stream
  float* SC  = X   + (size_t)T_ * D_;      // (H,S,S) raw scores for one batch
  float* MO  = SC  + (size_t)H_ * S_ * S_; // (T,D) moe accumulation
  float* CB  = MO  + (size_t)T_ * D_;      // (T,E)
  float* USG = CB  + (size_t)T_ * E_;      // (E)
  float* AUX = USG + E_;                   // 1 (+7 pad)
  float* PLD = AUX + 8;                    // (B,D)
  float* PT  = PLD + B_ * D_;              // (B,D)
  float* R1  = PT  + B_ * D_;              // (B,D2)
  float* FEND = R1 + B_ * D2_;
  _Float16* Hh16  = (_Float16*)FEND;                  // (T,D)
  _Float16* QKV16 = Hh16  + (size_t)T_ * D_;          // (T,3D)
  _Float16* CTX16 = QKV16 + (size_t)T_ * D3_;         // (T,D)
  _Float16* FO16  = CTX16 + (size_t)T_ * D_;          // (T,D) final LN out
  _Float16* P16   = FO16  + (size_t)T_ * D_;          // (H,S,S) softmax probs (per b)
  _Float16* VT16  = P16   + (size_t)H_ * S_ * S_;     // (B,H,64,S)
  _Float16* MID16 = VT16  + (size_t)B_ * H_ * DH_ * S_; // (T,I)
  _Float16* WQ    = MID16 + (size_t)T_ * I_;          // (3D,D)
  _Float16* WO    = WQ    + (size_t)D3_ * D_;         // (D,D)
  _Float16* W1T   = WO    + (size_t)D_ * D_;          // (I,D)
  _Float16* W2T   = W1T   + (size_t)I_ * D_;          // (D,I)

  const dim3 blk(256);
  k_copy<<<dim3((T_ * D_ + 255) / 256), blk, 0, stream>>>(X, emb, (long)T_ * D_);
  k_fill<<<dim3(1), blk, 0, stream>>>(AUX, 1, 0.f);

  for (int l = 0; l < L_; ++l) {
    // h = LN1(x) -> f16
    k_layernorm<float, _Float16><<<dim3(T_), blk, 0, stream>>>(
        X, nullptr, nullptr, ln1_w + l * D_, ln1_b + l * D_, Hh16);
    // qkv = h @ in_w.T + in_b  (f16 out)
    k_cvt<<<dim3((D3_ * D_ + 255) / 256), blk, 0, stream>>>(
        in_w + (size_t)l * D3_ * D_, WQ, (long)D3_ * D_);
    k_gemm<4, 0, 0><<<dim3(D3_ / 128, T_ / 128, 1), blk, 0, stream>>>(
        Hh16, D_, 0, WQ, D_, 0, QKV16, D3_, 0,
        in_b + (size_t)l * D3_, nullptr, nullptr, 0, D_);
    // V^T for P@V
    k_vt<<<dim3((B_ * H_ * DH_ * S_) / 256), blk, 0, stream>>>(QKV16, VT16);
    // attention per batch b, heads batched in grid.z
    for (int b = 0; b < B_; ++b) {
      const _Float16* qb = QKV16 + (size_t)b * S_ * D3_;
      k_gemm<4, 0, 3><<<dim3(S_ / 128, S_ / 128, H_), blk, 0, stream>>>(
          qb, D3_, DH_, qb + D_, D3_, DH_,
          SC, S_, (long)S_ * S_, nullptr, nullptr, nullptr, 0, DH_);
      k_softmax1024<<<dim3(H_ * S_), blk, 0, stream>>>(SC, P16, 0.125f);
      k_gemm<2, 0, 0><<<dim3(1, S_ / 128, H_), blk, 0, stream>>>(
          P16, S_, (long)S_ * S_,
          VT16 + (size_t)b * H_ * DH_ * S_, S_, (long)DH_ * S_,
          CTX16 + (size_t)b * S_ * D_, D_, DH_,
          nullptr, nullptr, nullptr, 0, S_);
    }
    // x = x + ctx @ out_w.T + out_b  (f32)
    k_cvt<<<dim3((D_ * D_ + 255) / 256), blk, 0, stream>>>(
        out_w + (size_t)l * D_ * D_, WO, (long)D_ * D_);
    k_gemm<4, 0, 1><<<dim3(D_ / 128, T_ / 128, 1), blk, 0, stream>>>(
        CTX16, D_, 0, WO, D_, 0, X, D_, 0,
        out_b + (size_t)l * D_, X, nullptr, 0, D_);
    // h = LN2(x) -> f16
    k_layernorm<float, _Float16><<<dim3(T_), blk, 0, stream>>>(
        X, nullptr, nullptr, ln2_w + l * D_, ln2_b + l * D_, Hh16);
    // gating
    k_fill<<<dim3(1), blk, 0, stream>>>(USG, E_, 0.f);
    k_gate<<<dim3(T_ / 256), blk, 0, stream>>>(
        Hh16, gate_w + (size_t)l * E_ * D_, CB, USG);
    k_aux<<<dim3(1), dim3(1), 0, stream>>>(USG, AUX);
    // MoE experts
    k_fill<<<dim3((T_ * D_ + 255) / 256), blk, 0, stream>>>(MO, (long)T_ * D_, 0.f);
    for (int e = 0; e < E_; ++e) {
      const size_t we = (size_t)(l * E_ + e);
      // e_w1 is (D,I): transpose-convert to (I,D)
      k_cvt_t<<<dim3((D_ * I_ + 255) / 256), blk, 0, stream>>>(
          e_w1 + we * D_ * I_, W1T, D_, I_);
      k_gemm<4, 1, 0><<<dim3(I_ / 128, T_ / 128, 1), blk, 0, stream>>>(
          Hh16, D_, 0, W1T, D_, 0, MID16, I_, 0,
          e_b1 + we * I_, nullptr, nullptr, 0, D_);
      // e_w2 is (I,D): transpose-convert to (D,I)
      k_cvt_t<<<dim3((D_ * I_ + 255) / 256), blk, 0, stream>>>(
          e_w2 + we * I_ * D_, W2T, I_, D_);
      k_gemm<4, 0, 2><<<dim3(D_ / 128, T_ / 128, 1), blk, 0, stream>>>(
          MID16, I_, 0, W2T, I_, 0, MO, D_, 0,
          e_b2 + we * D_, nullptr, CB + e, E_, I_);
    }
    // x = LN_m(h + moe_out * mask) -> f32
    k_layernorm<_Float16, float><<<dim3(T_), blk, 0, stream>>>(
        Hh16, MO, amask, mln_w + l * D_, mln_b + l * D_, X);
  }

  // head
  k_layernorm<float, _Float16><<<dim3(T_), blk, 0, stream>>>(
      X, nullptr, nullptr, fln_w, fln_b, FO16);
  k_pool<<<dim3(D_ / 256, B_), blk, 0, stream>>>(FO16, amask, PLD);
  k_small_fc<2><<<dim3((B_ * D_ + 255) / 256), blk, 0, stream>>>(
      PLD, pool_w, pool_b, PT, D_, D_, B_ * D_);
  k_small_fc<0><<<dim3(1), dim3(64), 0, stream>>>(
      PT, cls_w, cls_b, out, C_, D_, B_ * C_);           // logits -> out[0..15]
  k_small_fc<3><<<dim3((B_ * D2_ + 255) / 256), blk, 0, stream>>>(
      PT, cf_w1, cf_b1, R1, D2_, D_, B_ * D2_);
  k_small_fc<4><<<dim3(1), dim3(32), 0, stream>>>(
      R1, cf_w2, cf_b2, out + 17, 1, D2_, B_);           // conf -> out[17..20]
  k_write_aux<<<dim3(1), dim3(1), 0, stream>>>(AUX, out + 16); // aux -> out[16]
}
